// KNRMModel_82197084111644
// MI455X (gfx1250) — matrix operand order repaired and checked by reading, not compiled, not run on hardware
//
#include <hip/hip_runtime.h>
#include <hip/hip_bf16.h>
#include <math.h>

// ---------------- problem constants ----------------
#define NB   128   // batch
#define NC   5     // candidates
#define NH   50    // clicked history
#define NL   32    // title length
#define ND   300   // embedding dim
#define NDP  320   // padded K for WMMA (10 x 32)
#define NKN  20    // gaussian kernels

typedef _Float16 v8h  __attribute__((ext_vector_type(8)));
typedef _Float16 v16h __attribute__((ext_vector_type(16)));
typedef float    v8f  __attribute__((ext_vector_type(8)));

union V16 { v16h v; v8h h[2]; };

// ---------------------------------------------------------------------------
// Kernel 1: gather embedding rows, L2-normalize (clamp eps), convert to f16,
// pad K 300 -> 320 with zeros. One wave32 per row (coalesced over D).
// ---------------------------------------------------------------------------
__global__ void knrm_gather_norm_f16(const int* __restrict__ tok,
                                     const float* __restrict__ emb,
                                     _Float16* __restrict__ out,
                                     int nrows) {
  int wave = blockIdx.x * (blockDim.x >> 5) + (threadIdx.x >> 5);
  int lane = threadIdx.x & 31;
  if (wave >= nrows) return;

  const float* __restrict__ row = emb + (size_t)tok[wave] * ND;

  float vals[10];
  float ss = 0.0f;
  int n = 0;
  for (int d = lane; d < ND; d += 32) {
    float v = row[d];
    vals[n++] = v;
    ss += v * v;
  }
  // wave32 butterfly reduction of sum-of-squares
  #pragma unroll
  for (int off = 16; off > 0; off >>= 1)
    ss += __shfl_xor(ss, off, 32);

  float rn = 1.0f / fmaxf(sqrtf(ss), 1e-8f);

  _Float16* __restrict__ o = out + (size_t)wave * NDP;
  n = 0;
  for (int d = lane; d < ND; d += 32)
    o[d] = (_Float16)(vals[n++] * rn);
  for (int d = ND + lane; d < NDP; d += 32)
    o[d] = (_Float16)0.0f;
}

// ---------------------------------------------------------------------------
// Kernel 2: per (b,c,h): sim(32x32) = cn(32x320) @ hn(32x320)^T via
// v_wmma_f32_16x16x32_f16, sim tile kept in LDS, fused gaussian kernel
// pooling -> pv[b,c,h,0..19].  128 threads = 4 wave32s, one 16x16 tile each.
// ---------------------------------------------------------------------------
__global__ void __launch_bounds__(128)
knrm_sim_pool(const _Float16* __restrict__ ce,
              const _Float16* __restrict__ he,
              float* __restrict__ pv) {
  const int blk = blockIdx.x;          // (b*NC + c)*NH + h
  const int h   = blk % NH;
  const int bc  = blk / NH;
  const int b   = bc / NC;

  const _Float16* __restrict__ A  = ce + (size_t)bc * NL * NDP;              // 32 x 320 (rows i)
  const _Float16* __restrict__ Bm = he + ((size_t)b * NH + h) * NL * NDP;    // 32 x 320 (rows j)

  const int tid  = threadIdx.x;
  const int wid  = tid >> 5;           // wave 0..3
  const int lane = tid & 31;
  const int hl   = lane >> 4;          // lane half (0/1)
  const int m    = lane & 15;
  const int mt   = wid >> 1;           // row tile 0/1
  const int nt   = wid & 1;            // col tile 0/1

  // A fragment rows: lane -> row m of the tile; K chunks per ISA A layout
  const _Float16* __restrict__ arow = A  + (size_t)(mt * 16 + m) * NDP;
  // B fragment: lane -> column n (= hn row j); K chunks in the same order as A
  const _Float16* __restrict__ brow = Bm + (size_t)(nt * 16 + m) * NDP;

  v8f acc = {};
  #pragma unroll
  for (int ks = 0; ks < NDP; ks += 32) {
    V16 a, bb;
    // A (16x32 f16): lanes 0-15: K=[ks,ks+8) & [ks+16,ks+24); lanes 16-31: +8
    a.h[0] = *(const v8h*)(arow + ks + 8 * hl);
    a.h[1] = *(const v8h*)(arow + ks + 16 + 8 * hl);
    // B (32x16 f16): same K order as A: lanes 0-15: K=[ks,ks+8) & [ks+16,ks+24); lanes 16-31: +8
    bb.h[0] = *(const v8h*)(brow + ks + 8 * hl);
    bb.h[1] = *(const v8h*)(brow + ks + 16 + 8 * hl);
    acc = __builtin_amdgcn_wmma_f32_16x16x32_f16(
        /*neg_a=*/false, a.v, /*neg_b=*/false, bb.v,
        /*c_mod=*/(short)0, acc, /*reuse_a=*/false, /*reuse_b=*/false);
  }

  __shared__ float sim[NL][NL + 1];    // +1 pad: no LDS bank conflicts
  __shared__ float pvl[NKN];

  // C/D layout: VGPR r -> row r (lanes 0-15) / row r+8 (lanes 16-31), col = lane&15
  #pragma unroll
  for (int r = 0; r < 8; ++r)
    sim[mt * 16 + hl * 8 + r][nt * 16 + m] = acc[r];

  if (tid < NKN) pvl[tid] = 0.0f;
  __syncthreads();

  // pooled[i][k] = sum_j exp(-(s-mu_k)^2/(2 sig_k^2)); pv[k] += log(clip(pooled))
  for (int idx = tid; idx < NL * NKN; idx += 128) {
    const int i = idx / NKN;
    const int k = idx - i * NKN;
    const float mu = 0.1f * (float)k - 0.9f;
    const float c2 = (k == NKN - 1) ? 500000.0f : 50.0f;  // 1/(2*sigma^2)
    float s = 0.0f;
    #pragma unroll 8
    for (int j = 0; j < NL; ++j) {
      float d = sim[i][j] - mu;
      s += __expf(-d * d * c2);
    }
    atomicAdd(&pvl[k], __logf(fmaxf(s, 1e-10f)));   // ds_add_f32
  }
  __syncthreads();

  if (tid < NKN)
    pv[(size_t)blk * NKN + tid] = pvl[tid];
}

// ---------------------------------------------------------------------------
// Kernel 3: score[b][c] = pv[b,c,:] . w + bias ; out = log_softmax over C
// ---------------------------------------------------------------------------
__global__ void knrm_score(const float* __restrict__ pv,
                           const float* __restrict__ w,
                           const float* __restrict__ bias,
                           float* __restrict__ out) {
  const int b = blockIdx.x;
  const int tid = threadIdx.x;
  __shared__ float sc[NC];

  if (tid < NC) {
    const float* __restrict__ p = pv + ((size_t)b * NC + tid) * (NH * NKN);
    float acc = bias[0];
    for (int j = 0; j < NH * NKN; ++j)
      acc = fmaf(p[j], w[j], acc);
    sc[tid] = acc;
  }
  __syncthreads();
  if (tid == 0) {
    float mx = sc[0];
    #pragma unroll
    for (int c = 1; c < NC; ++c) mx = fmaxf(mx, sc[c]);
    float s = 0.0f;
    #pragma unroll
    for (int c = 0; c < NC; ++c) s += __expf(sc[c] - mx);
    float lse = mx + __logf(s);
    #pragma unroll
    for (int c = 0; c < NC; ++c) out[b * NC + c] = sc[c] - lse;
  }
}

// ---------------------------------------------------------------------------
extern "C" void kernel_launch(void* const* d_in, const int* in_sizes, int n_in,
                              void* d_out, int out_size, void* d_ws, size_t ws_size,
                              hipStream_t stream) {
  const int*   cand = (const int*)d_in[0];    // [B,C,L]
  const int*   clk  = (const int*)d_in[1];    // [B,H,L]
  const float* emb  = (const float*)d_in[2];  // [VOCAB,300]
  const float* w    = (const float*)d_in[3];  // [1, H*KN]
  const float* bia  = (const float*)d_in[4];  // [1]
  float*       out  = (float*)d_out;          // [B,C]

  // workspace layout (all 16B+ aligned)
  _Float16* ce = (_Float16*)d_ws;                               // 13.1 MB
  _Float16* he = ce + (size_t)NB * NC * NL * NDP;               // 131 MB
  float*    pv = (float*)(he + (size_t)NB * NH * NL * NDP);     // 2.56 MB

  const int candRows = NB * NC * NL;   // 20480
  const int clkRows  = NB * NH * NL;   // 204800

  // 8 wave32s per block, one row per wave
  knrm_gather_norm_f16<<<(candRows + 7) / 8, 256, 0, stream>>>(cand, emb, ce, candRows);
  knrm_gather_norm_f16<<<(clkRows  + 7) / 8, 256, 0, stream>>>(clk,  emb, he, clkRows);

  knrm_sim_pool<<<NB * NC * NH, 128, 0, stream>>>(ce, he, pv);

  knrm_score<<<NB, 32, 0, stream>>>(pv, w, bia, out);
}
